// AnyPrecisionLinear_21019569946790
// MI455X (gfx1250) — matrix operand order repaired and checked by reading, not compiled, not run on hardware
//
#include <hip/hip_runtime.h>

// CDNA5 / gfx1250 wave32 WMMA types
typedef __attribute__((ext_vector_type(16))) _Float16 v16h;
typedef __attribute__((ext_vector_type(8)))  _Float16 v8h;
typedef __attribute__((ext_vector_type(8)))  float    v8f;

#define TM 128
#define TN 128
#define TK 64
#define LDP (TK + 8)   // padded LDS row stride in halves (144B = 9*16B, keeps 16B alignment)

// Async global->LDS copy (CDNA5): no VGPR staging, tracked by ASYNCcnt.
// vdst = per-lane LDS byte address (addrspace(3) ptr in one VGPR),
// vaddr = per-lane 64-bit global address (VGPR pair), saddr = off.
__device__ __forceinline__ void async_load_b128(void* lds_dst, const void* gsrc) {
    asm volatile("global_load_async_to_lds_b128 %0, %1, off"
                 :
                 : "v"((__attribute__((address_space(3))) void*)lds_dst), "v"(gsrc)
                 : "memory");
}
__device__ __forceinline__ void wait_async0() {
    asm volatile("s_wait_asynccnt 0x0" ::: "memory");
}

// 4-bit bitplane dequant + f16 WMMA GEMM:  out[m,n] = sum_k x[m,k]*W[n,k] + bias[n]
// W[n, widx*32+j] = lut[n, code]; code = plane bits, plane 0 = MSB, bit j at shift 31-j.
__global__ void __launch_bounds__(256)
apl4_wmma_f16_kernel(const _Float16* __restrict__ x,
                     const unsigned int* __restrict__ qw,
                     const _Float16* __restrict__ lut,
                     const _Float16* __restrict__ bias,
                     float* __restrict__ out,
                     int M, int N, int K, int Kwords) {
    __shared__ _Float16 sA[2][TM * LDP];   // 2 x 18 KB : X tiles (double buffered)
    __shared__ _Float16 sB[2][TN * LDP];   // 2 x 18 KB : dequantized W tiles
    __shared__ _Float16 sLut[TN * 16];     //      4 KB : per-row codebooks

    const int tid  = threadIdx.x;
    const int lane = tid & 31;
    const int w    = tid >> 5;        // wave 0..7
    const int wm   = w & 3;           // 4 waves along M: offset wm*32
    const int wn   = w >> 2;          // 2 waves along N: offset wn*64
    const int n0   = blockIdx.x * TN;
    const int m0   = blockIdx.y * TM;

    // stage LUT tile: 128 rows x 16 fp16; each thread copies 8 halves (16B)
    {
        int r = tid >> 1;
        int c = (tid & 1) * 8;
        *(v8h*)&sLut[r * 16 + c] = *(const v8h*)&lut[(size_t)(n0 + r) * 16 + c];
    }

    v8f acc[2][4];
    #pragma unroll
    for (int mi = 0; mi < 2; ++mi)
        #pragma unroll
        for (int ni = 0; ni < 4; ++ni)
            acc[mi][ni] = (v8f){0.f, 0.f, 0.f, 0.f, 0.f, 0.f, 0.f, 0.f};

    // per-thread dequant ownership: row o_local = tid/2, word wsel = tid&1 (TK=64 -> 2 words/row)
    const int o_local = tid >> 1;
    const int wsel    = tid & 1;
    const size_t qrow         = (size_t)(n0 + o_local) * (size_t)Kwords;
    const size_t plane_stride = (size_t)N * (size_t)Kwords;

    __syncthreads();   // sLut visible before first dequant gather

    // ---- stage one K-step tile (async A copy + B dequant) into buffer b ----
    auto stage_tile = [&](int kk, int b) {
        // X tile: 128 x 64 halves; 1024 async 16B chunks, 4 per thread
        _Float16* A = &sA[b][0];
        #pragma unroll
        for (int i = 0; i < 4; ++i) {
            int c   = tid + 256 * i;
            int row = c >> 3;
            int col = (c & 7) * 8;
            async_load_b128(&A[row * LDP + col],
                            &x[(size_t)(m0 + row) * K + kk + col]);
        }
        // dequant one 32-weight word into the B tile
        int widx = (kk >> 5) + wsel;
        unsigned int p0 = qw[0 * plane_stride + qrow + widx];
        unsigned int p1 = qw[1 * plane_stride + qrow + widx];
        unsigned int p2 = qw[2 * plane_stride + qrow + widx];
        unsigned int p3 = qw[3 * plane_stride + qrow + widx];
        _Float16*       dst  = &sB[b][o_local * LDP + wsel * 32];
        const _Float16* lrow = &sLut[o_local * 16];
        #pragma unroll
        for (int j = 0; j < 32; ++j) {
            int sh   = 31 - j;                       // MSB-first within word
            int code = (int)((((p0 >> sh) & 1u) << 3) | (((p1 >> sh) & 1u) << 2) |
                             (((p2 >> sh) & 1u) << 1) |  ((p3 >> sh) & 1u));
            dst[j] = lrow[code];
        }
        // prefetch bitplane words for the K-step after next
        if (kk + TK < K)
            __builtin_prefetch(&qw[qrow + ((kk + TK) >> 5) + wsel], 0, 3);
    };

    // ---- WMMA over one staged buffer ----
    auto mma_tile = [&](int b) {
        const _Float16* A = &sA[b][0];
        const _Float16* B = &sB[b][0];
        #pragma unroll
        for (int kc = 0; kc < TK; kc += 32) {
            // A frags 16x32 f16: lanes 0-15 K {0-7,16-23}, lanes 16-31 K {8-15,24-31}
            v16h af[2];
            #pragma unroll
            for (int mi = 0; mi < 2; ++mi) {
                int row = wm * 32 + mi * 16 + (lane & 15);
                int kb  = kc + ((lane >> 4) << 3);            // 0 or 8
                v8h lo = *(v8h*)&A[row * LDP + kb];
                v8h hi = *(v8h*)&A[row * LDP + kb + 16];
                v16h a;
                #pragma unroll
                for (int e = 0; e < 8; ++e) { a[e] = lo[e]; a[e + 8] = hi[e]; }
                af[mi] = a;
            }
            // B frags 32x16 f16: lane = N column; lanes 0-15 K {0-7,16-23}, lanes 16-31 K {8-15,24-31}
            v16h bf[4];
            #pragma unroll
            for (int ni = 0; ni < 4; ++ni) {
                int o  = wn * 64 + ni * 16 + (lane & 15);
                int kb = kc + ((lane >> 4) << 3);             // 0 or 8
                v8h lo = *(v8h*)&B[o * LDP + kb];
                v8h hi = *(v8h*)&B[o * LDP + kb + 16];
                v16h bfr;
                #pragma unroll
                for (int e = 0; e < 8; ++e) { bfr[e] = lo[e]; bfr[e + 8] = hi[e]; }
                bf[ni] = bfr;
            }
            #pragma unroll
            for (int mi = 0; mi < 2; ++mi)
                #pragma unroll
                for (int ni = 0; ni < 4; ++ni)
                    acc[mi][ni] = __builtin_amdgcn_wmma_f32_16x16x32_f16(
                        false, af[mi], false, bf[ni], (short)0, acc[mi][ni], false, false);
        }
    };

    // ---- software pipeline: stage s+1 while computing s; 1 barrier per K-step ----
    int buf = 0;
    stage_tile(0, buf);
    wait_async0();
    __syncthreads();

    const int nsteps = K / TK;
    for (int s = 0; s < nsteps; ++s) {
        if (s + 1 < nsteps)
            stage_tile((s + 1) * TK, buf ^ 1);   // writes buf^1: no conflict with readers of buf
        mma_tile(buf);
        wait_async0();                           // async LDS writes (buf^1) landed
        __syncthreads();                         // + dscnt waited by barrier lowering
        buf ^= 1;
    }

    // ---- epilogue: f32 C/D layout (VGPR r: M=r for lanes 0-15, M=r+8 for lanes 16-31)
    #pragma unroll
    for (int mi = 0; mi < 2; ++mi) {
        int mrow0 = m0 + wm * 32 + mi * 16 + ((lane >> 4) << 3);
        #pragma unroll
        for (int ni = 0; ni < 4; ++ni) {
            int   ncol = n0 + wn * 64 + ni * 16 + (lane & 15);
            float bv   = (float)bias[ncol];
            #pragma unroll
            for (int r = 0; r < 8; ++r)
                out[(size_t)(mrow0 + r) * N + ncol] = acc[mi][ni][r] + bv;
        }
    }
}

extern "C" void kernel_launch(void* const* d_in, const int* in_sizes, int n_in,
                              void* d_out, int out_size, void* d_ws, size_t ws_size,
                              hipStream_t stream) {
    const _Float16*     x    = (const _Float16*)d_in[0];
    const unsigned int* qw   = (const unsigned int*)d_in[1];
    const _Float16*     lut  = (const _Float16*)d_in[2];
    const _Float16*     bias = (const _Float16*)d_in[3];
    float*              out  = (float*)d_out;

    const int N      = in_sizes[3];                 // 11008 output channels (bias count)
    const int Kwords = in_sizes[1] / (4 * N);       // qweight = (4, N, K/32) -> 128
    const int K      = Kwords * 32;                 // 4096
    const int M      = in_sizes[0] / K;             // 8192 tokens

    dim3 grid(N / TN, M / TM);                      // 86 x 64 blocks
    dim3 block(256);
    apl4_wmma_f16_kernel<<<grid, block, 0, stream>>>(x, qw, lut, bias, out, M, N, K, Kwords);
}
